// GAT_4947802325596
// MI455X (gfx1250) — matrix operand order repaired and checked by reading, not compiled, not run on hardware
//
#include <hip/hip_runtime.h>

#define NEG_SLOPE 0.2f

typedef __attribute__((ext_vector_type(16))) __bf16        v16bf;
typedef __attribute__((ext_vector_type(8)))  float         v8f;
typedef __attribute__((ext_vector_type(8)))  unsigned int  v8u;

// ---------------- small helpers ----------------

__device__ __forceinline__ unsigned short f2bf(float f) {
  unsigned int u = __float_as_uint(f);
  u += 0x7FFFu + ((u >> 16) & 1u);        // round-to-nearest-even
  return (unsigned short)(u >> 16);
}

__device__ __forceinline__ float lrelu(float x) { return x > 0.f ? x : NEG_SLOPE * x; }

// monotonic float<->uint mapping so we can use integer atomicMax for float max
__device__ __forceinline__ unsigned int fenc(float x) {
  unsigned int u = __float_as_uint(x);
  return (u & 0x80000000u) ? ~u : (u | 0x80000000u);
}
__device__ __forceinline__ float fdec(unsigned int u) {
  unsigned int f = (u & 0x80000000u) ? (u & 0x7FFFFFFFu) : ~u;
  return __uint_as_float(f);
}

// ---------------- conversion / fill kernels ----------------

__global__ void k_f32_to_bf16(const float* __restrict__ in,
                              unsigned short* __restrict__ out, long n) {
  long i = (long)blockIdx.x * blockDim.x + threadIdx.x;
  if (i < n) out[i] = f2bf(in[i]);
}

// W[K][C] (row-major) -> Wt[C][K] bf16, so B-fragment loads are contiguous in K
__global__ void k_wT_bf16(const float* __restrict__ W,
                          unsigned short* __restrict__ wt, int K, int C) {
  long i = (long)blockIdx.x * blockDim.x + threadIdx.x;
  if (i >= (long)K * C) return;
  int k = (int)(i / C), n = (int)(i % C);
  wt[(size_t)n * K + k] = f2bf(W[i]);
}

__global__ void k_fill_u32(unsigned int* __restrict__ p, unsigned int v, long n) {
  long i = (long)blockIdx.x * blockDim.x + threadIdx.x;
  if (i < n) p[i] = v;
}

// ---------------- WMMA GEMM strip kernel ----------------
// C[M, NTN*16] = A[M,K](bf16) x W[K, NTN*16](bf16, given transposed as Bt[N,K]).
// One wave computes a full 16-row x Ncol strip:
//  - the A fragment is loaded once per K-step and reused across all NTN column tiles
//  - all NTN B fragments are materialized first so the loads clause together and the
//    NTN WMMAs issue back-to-back
//  - the D tiles are staged through LDS so global stores are coalesced b128

template <int NTN>
__global__ __launch_bounds__(256) void k_wmma_gemm_strip(
    const unsigned short* __restrict__ A,   // [M x K] row-major bf16
    const unsigned short* __restrict__ Bt,  // [Ncol x K] row-major bf16 (== W^T)
    float* __restrict__ C, int M, int K)
{
  const int Ncol = NTN * 16;
  __shared__ float cbuf[8][16][NTN * 16];          // 8 KB per wave (64 KB for NTN=8)

  const int lane = threadIdx.x & 31;
  const int wave = threadIdx.x >> 5;
  const long tilesM = (long)((M + 15) >> 4);
  const long t = (long)blockIdx.x * 8 + wave;      // wave-uniform row-tile index
  if (t >= tilesM) return;                         // whole wave exits: EXEC all-1 for WMMA
  const int tm = (int)t;

  const int l15  = lane & 15;
  const int half = lane >> 4;                      // 0: lanes 0-15, 1: lanes 16-31
  int mrow = tm * 16 + l15; if (mrow >= M) mrow = M - 1;   // clamp (stores guarded)

  const unsigned int* arow = (const unsigned int*)(A + (size_t)mrow * K);

  const v8f vzero = {};
  v8f acc[NTN];
#pragma unroll
  for (int tn = 0; tn < NTN; ++tn) acc[tn] = vzero;

  for (int kk = 0; kk < K; kk += 32) {
    // A fragment (16x32 bf16): lane half 0 holds K {0..7,16..23}, half 1 holds K {8..15,24..31}
    const int ka = (kk + half * 8) >> 1;           // u32 units (2 bf16 per u32)
    v8u au;
    au[0] = arow[ka + 0];  au[1] = arow[ka + 1];
    au[2] = arow[ka + 2];  au[3] = arow[ka + 3];
    au[4] = arow[ka + 8];  au[5] = arow[ka + 9];
    au[6] = arow[ka + 10]; au[7] = arow[ka + 11];
    const v16bf a = __builtin_bit_cast(v16bf, au);

    // all B fragments first (lets the loads clause), then WMMAs back-to-back
    const int kb = (kk + half * 8) >> 1;
    v8u bu[NTN];
#pragma unroll
    for (int tn = 0; tn < NTN; ++tn) {
      const unsigned int* brow =
          (const unsigned int*)(Bt + (size_t)(tn * 16 + l15) * K);
#pragma unroll
      for (int v = 0; v < 8; ++v) bu[tn][v] = brow[kb + v + (v >> 2) * 4];
    }
#pragma unroll
    for (int tn = 0; tn < NTN; ++tn)
      acc[tn] = __builtin_amdgcn_wmma_f32_16x16x32_bf16(
          false, a, false, __builtin_bit_cast(v16bf, bu[tn]),
          (short)0, acc[tn], false, false);
  }

  // Stage D tiles in this wave's private LDS slab.
  // D layout: lane col = l15 (+tile offset); VGPR v holds row v (+8 for lanes 16-31).
  // Same-wave DS ops are in-order, so no barrier is required.
#pragma unroll
  for (int tn = 0; tn < NTN; ++tn) {
    const int ncol = tn * 16 + l15;
#pragma unroll
    for (int v = 0; v < 8; ++v)
      cbuf[wave][half * 8 + v][ncol] = acc[tn][v];
  }

  // Read back row-major, emit coalesced 128-bit global stores.
  const int iters = (16 * Ncol) / 128;             // 128 floats per wave-iteration
#pragma unroll
  for (int j = 0; j < iters; ++j) {
    const int idx = j * 128 + lane * 4;
    const int r = idx / Ncol, c = idx % Ncol;
    const int grow = tm * 16 + r;
    if (grow < M) {
      const float4 val = *(const float4*)&cbuf[wave][r][c];
      *(float4*)&C[(size_t)grow * Ncol + c] = val;
    }
  }
}

// ---------------- attention scalar kernels ----------------

// el[n,h] = sum_d ft[n,h,d]*al[h,d] ; er likewise
__global__ void k_eler(const float* __restrict__ ft, const float* __restrict__ al,
                       const float* __restrict__ ar, float* __restrict__ el,
                       float* __restrict__ er, int Nn, int H, int D) {
  long i = (long)blockIdx.x * blockDim.x + threadIdx.x;
  if (i >= (long)Nn * H) return;
  int h = (int)(i % H);
  const float* f = ft + i * (long)D;     // (n*H + h)*D == n*(H*D) + h*D
  float sl = 0.f, sr = 0.f;
  for (int d = 0; d < D; ++d) { float x = f[d]; sl += x * al[h * D + d]; sr += x * ar[h * D + d]; }
  el[i] = sl; er[i] = sr;
}

// pass 1: per-(edge,head) leaky-relu attention logit, segment max via integer atomicMax
__global__ void k_edge_max(const int* __restrict__ src, const int* __restrict__ dst,
                           const float* __restrict__ el, const float* __restrict__ er,
                           unsigned int* __restrict__ emax, long EH, int H) {
  long i = (long)blockIdx.x * blockDim.x + threadIdx.x;
  if (i >= EH) return;
  long e = i / H; int h = (int)(i % H);
  int s = src[e], d = dst[e];
  float x = lrelu(el[(size_t)s * H + h] + er[(size_t)d * H + h]);
  atomicMax(&emax[(size_t)d * H + h], fenc(x));
}

// pass 2: ex = exp(e - max); stash per-edge ex; segment-sum denominator
__global__ void k_edge_expsum(const int* __restrict__ src, const int* __restrict__ dst,
                              const float* __restrict__ el, const float* __restrict__ er,
                              const unsigned int* __restrict__ emax,
                              float* __restrict__ exb, float* __restrict__ denom,
                              long EH, int H) {
  long i = (long)blockIdx.x * blockDim.x + threadIdx.x;
  if (i >= EH) return;
  long e = i / H; int h = (int)(i % H);
  int s = src[e], d = dst[e];
  float x = lrelu(el[(size_t)s * H + h] + er[(size_t)d * H + h]);
  float m = fdec(emax[(size_t)d * H + h]);
  float ex = __expf(x - m);
  exb[i] = ex;
  atomicAdd(&denom[(size_t)d * H + h], ex);
}

// pass 3: acc[dst, c] += ft[src, c] * alpha(e, head(c))   (L2-resident scatter-add)
__global__ void k_edge_agg(const int* __restrict__ src, const int* __restrict__ dst,
                           const float* __restrict__ ft, const float* __restrict__ exb,
                           const float* __restrict__ denom, float* __restrict__ acc,
                           long total, int H, int D) {
  long i = (long)blockIdx.x * blockDim.x + threadIdx.x;
  if (i >= total) return;
  const int C = H * D;
  long e = i / C; int c = (int)(i % C); int h = c / D;
  int s = src[e], d = dst[e];
  float alpha = exb[e * H + h] / denom[(size_t)d * H + h];
  atomicAdd(&acc[(size_t)d * C + c], ft[(size_t)s * C + c] * alpha);
}

__global__ void k_bias_act(const float* __restrict__ acc, const float* __restrict__ bias,
                           float* __restrict__ out, long total, int C, int relu) {
  long i = (long)blockIdx.x * blockDim.x + threadIdx.x;
  if (i >= total) return;
  int c = (int)(i % C);
  float x = acc[i] + bias[c];
  if (relu) x = fmaxf(x, 0.f);
  out[i] = x;
}

// ---------------- host orchestration ----------------

static inline int nblk(long n) { return (int)((n + 255) / 256); }

static void run_layer(const float* hin, int Cin, const float* W, const float* al,
                      const float* ar, const float* bias, int H, int D,
                      const int* src, const int* dst, int Nn, long Ee,
                      unsigned short* hbf, unsigned short* wbfT,
                      float* ft, float* acc, float* hout, int relu,
                      float* el, float* er, unsigned int* emax, float* denom, float* exb,
                      hipStream_t stream)
{
  const int C = H * D;
  const long nIn = (long)Nn * Cin;
  k_f32_to_bf16<<<nblk(nIn), 256, 0, stream>>>(hin, hbf, nIn);
  k_wT_bf16<<<nblk((long)Cin * C), 256, 0, stream>>>(W, wbfT, Cin, C);
  {
    long tilesM = (long)((Nn + 15) / 16);
    int  blocks = (int)((tilesM + 7) / 8);         // 8 waves (16-row strips) per block
    if (C == 128)
      k_wmma_gemm_strip<8><<<blocks, 256, 0, stream>>>(hbf, wbfT, ft, Nn, Cin);
    else
      k_wmma_gemm_strip<4><<<blocks, 256, 0, stream>>>(hbf, wbfT, ft, Nn, Cin);
  }
  k_eler<<<nblk((long)Nn * H), 256, 0, stream>>>(ft, al, ar, el, er, Nn, H, D);
  k_fill_u32<<<nblk((long)Nn * H), 256, 0, stream>>>(emax, 0u, (long)Nn * H);
  k_fill_u32<<<nblk((long)Nn * H), 256, 0, stream>>>((unsigned int*)denom, 0u, (long)Nn * H);
  k_fill_u32<<<nblk((long)Nn * C), 256, 0, stream>>>((unsigned int*)acc, 0u, (long)Nn * C);
  k_edge_max<<<nblk(Ee * H), 256, 0, stream>>>(src, dst, el, er, emax, Ee * H, H);
  k_edge_expsum<<<nblk(Ee * H), 256, 0, stream>>>(src, dst, el, er, emax, exb, denom, Ee * H, H);
  k_edge_agg<<<nblk(Ee * (long)C), 256, 0, stream>>>(src, dst, ft, exb, denom, acc, Ee * (long)C, H, D);
  k_bias_act<<<nblk((long)Nn * C), 256, 0, stream>>>(acc, bias, hout, (long)Nn * C, C, relu);
}

extern "C" void kernel_launch(void* const* d_in, const int* in_sizes, int n_in,
                              void* d_out, int out_size, void* d_ws, size_t ws_size,
                              hipStream_t stream)
{
  const float* features = (const float*)d_in[0];
  const int*   src      = (const int*)d_in[1];
  const int*   dst      = (const int*)d_in[2];
  const float* W0  = (const float*)d_in[3];
  const float* al0 = (const float*)d_in[4];
  const float* ar0 = (const float*)d_in[5];
  const float* b0  = (const float*)d_in[6];
  const float* W1  = (const float*)d_in[7];
  const float* al1 = (const float*)d_in[8];
  const float* ar1 = (const float*)d_in[9];
  const float* b1  = (const float*)d_in[10];
  const float* W2  = (const float*)d_in[11];
  const float* al2 = (const float*)d_in[12];
  const float* ar2 = (const float*)d_in[13];
  const float* b2  = (const float*)d_in[14];
  float* out = (float*)d_out;

  const int  Nn = in_sizes[0] / 128;
  const long Ee = (long)in_sizes[1];

  // workspace carve-up (256B aligned)
  char* ws = (char*)d_ws;
  size_t off = 0;
  auto carve = [&](size_t bytes) -> char* {
    char* p = ws + off;
    off += (bytes + 255) & ~(size_t)255;
    return p;
  };
  float*          bufX  = (float*)carve((size_t)Nn * 128 * sizeof(float));
  float*          bufY  = (float*)carve((size_t)Nn * 128 * sizeof(float));
  unsigned short* hbf   = (unsigned short*)carve((size_t)Nn * 128 * sizeof(unsigned short));
  unsigned short* wbfT  = (unsigned short*)carve((size_t)128 * 128 * sizeof(unsigned short));
  float*          el    = (float*)carve((size_t)Nn * 4 * sizeof(float));
  float*          er    = (float*)carve((size_t)Nn * 4 * sizeof(float));
  unsigned int*   emax  = (unsigned int*)carve((size_t)Nn * 4 * sizeof(unsigned int));
  float*          denom = (float*)carve((size_t)Nn * 4 * sizeof(float));
  float*          exb   = (float*)carve((size_t)Ee * 4 * sizeof(float));
  (void)ws_size; (void)n_in; (void)out_size;

  // layer 0: hin=features, ft->bufY, acc/out->bufX (relu)
  run_layer(features, 128, W0, al0, ar0, b0, /*H=*/4, /*D=*/32, src, dst, Nn, Ee,
            hbf, wbfT, /*ft*/bufY, /*acc*/bufX, /*hout*/bufX, /*relu*/1,
            el, er, emax, denom, exb, stream);
  // layer 1: hin=bufX (consumed into hbf before ft overwrite), ft->bufX, acc/out->bufY (relu)
  run_layer(bufX, 128, W1, al1, ar1, b1, 4, 32, src, dst, Nn, Ee,
            hbf, wbfT, bufX, bufY, bufY, 1,
            el, er, emax, denom, exb, stream);
  // layer 2: 1 head, 64 dims, no relu, write final result to d_out
  run_layer(bufY, 128, W2, al2, ar2, b2, 1, 64, src, dst, Nn, Ee,
            hbf, wbfT, bufY, bufX, out, 0,
            el, er, emax, denom, exb, stream);
}
